// ScaleDotSelfAttentionAbs_62045097558236
// MI455X (gfx1250) — matrix operand order repaired and checked by reading, not compiled, not run on hardware
//
#include <hip/hip_runtime.h>
#include <stdint.h>

// ---------------- problem dims ----------------
constexpr int HD = 1024;        // hidden
constexpr int SL = 2048;        // sequence
constexpr int NBATCH = 8;
constexpr int BSROWS = NBATCH * SL;   // 16384

// ---------------- wmma vector types ----------------
typedef __attribute__((ext_vector_type(16))) __bf16 v16bf;
typedef __attribute__((ext_vector_type(8)))  __bf16 v8bf;
typedef __attribute__((ext_vector_type(8)))  float  v8f;
typedef __attribute__((ext_vector_type(4)))  int    v4i;

// ---------------- bf16 helpers (native cvt on gfx1250) ----------------
__device__ __forceinline__ __bf16 f2bf(float f) { return (__bf16)f; }

__device__ __forceinline__ float waveMax(float v) {
  #pragma unroll
  for (int off = 16; off; off >>= 1) v = fmaxf(v, __shfl_xor(v, off, 32));
  return v;
}
__device__ __forceinline__ float waveSum(float v) {
  #pragma unroll
  for (int off = 16; off; off >>= 1) v += __shfl_xor(v, off, 32);
  return v;
}

// ---------------- CDNA5 async global->LDS copy (16B per lane) ----------------
#if __has_builtin(__builtin_amdgcn_global_load_async_to_lds_b128)
#define HAVE_ASYNC_LDS 1
#else
#define HAVE_ASYNC_LDS 0
#endif

typedef __attribute__((address_space(1))) v4i gv4i_t;
typedef __attribute__((address_space(3))) v4i lv4i_t;

__device__ __forceinline__ void async_copy16(void* lds_dst, const void* gsrc) {
#if HAVE_ASYNC_LDS
  __builtin_amdgcn_global_load_async_to_lds_b128(
      (gv4i_t*)(gsrc), (lv4i_t*)(lds_dst), 0, 0);
#else
  *(v8bf*)lds_dst = *(const v8bf*)gsrc;
#endif
}

__device__ __forceinline__ void wait_async() {
#if __has_builtin(__builtin_amdgcn_s_wait_asynccnt)
  __builtin_amdgcn_s_wait_asynccnt(0);
#elif HAVE_ASYNC_LDS
  asm volatile("s_wait_asynccnt 0x0" ::: "memory");
#endif
}

// ---------------- weight transpose+convert: Wt[n][k] = bf16(W[k][n]) ----------------
__global__ __launch_bounds__(256)
void wtrans_kernel(const float* __restrict__ W, __bf16* __restrict__ Wt) {
  int idx = blockIdx.x * 256 + threadIdx.x;   // n*HD + k
  int n = idx >> 10;
  int k = idx & (HD - 1);
  Wt[idx] = f2bf(W[k * HD + n]);
}

// ---------------- generic bf16 WMMA GEMM ----------------
// C[m][n] = scale * sum_k A[m][k] * Bt[n][k]  (+bias[n]) (+addmat[m][n])
// 128x128 C tile per 256-thread block (8 waves), BK = 32.
// Wave w: m-block = (w&3)*32 (2 M-tiles), n-block = (w>>2)*64 (4 N-tiles).
__global__ __launch_bounds__(256)
void gemm_bf16_kernel(const void* __restrict__ Av, int a_is_f32, long lda, long bsA,
                      const __bf16* __restrict__ Btp, long ldb, long bsB,
                      const float* __restrict__ bias,
                      const float* __restrict__ addmat, long ld_add,
                      float scale,
                      void* __restrict__ Cv, int c_is_f32, long ldc, long bsC,
                      int c_trans, int trans_S,
                      int Ksize) {
  __shared__ __bf16 As[128][40];   // 80B row stride: 16B aligned, conflict-spread
  __shared__ __bf16 Bs[128][40];

  const int tid  = threadIdx.x;
  const int lane = tid & 31;
  const int wave = tid >> 5;
  const int hl   = lane >> 4;      // lane half (wave32)
  const int lrow = lane & 15;
  const int wm   = wave & 3;       // 32-row block
  const int wn   = wave >> 2;      // 64-col block

  const long z  = blockIdx.z;
  const float*  Af = (const float*)Av  + z * bsA;
  const __bf16* Ab = (const __bf16*)Av + z * bsA;
  const __bf16* Bp = Btp + z * bsB;

  const long m0 = (long)blockIdx.y * 128;
  const long n0 = (long)blockIdx.x * 128;

  v8f acc[2][4];
  #pragma unroll
  for (int i = 0; i < 2; ++i)
    #pragma unroll
    for (int j = 0; j < 4; ++j)
      #pragma unroll
      for (int r = 0; r < 8; ++r) acc[i][j][r] = 0.0f;

  for (int k0 = 0; k0 < Ksize; k0 += 32) {
    // ---- stage A tile (128 x 32) into LDS as bf16 ----
    if (a_is_f32) {
      #pragma unroll
      for (int i = 0; i < 4; ++i) {
        int c   = tid + 256 * i;        // 0..1023 float4 chunks
        int row = c >> 3;
        int col = (c & 7) * 4;
        const float4 v = *(const float4*)(Af + (m0 + row) * lda + k0 + col);
        __bf16* d = &As[row][col];
        d[0] = f2bf(v.x); d[1] = f2bf(v.y); d[2] = f2bf(v.z); d[3] = f2bf(v.w);
      }
      if (k0 + 32 < Ksize)
        __builtin_prefetch(Af + (m0 + (tid >> 1)) * lda + k0 + 32, 0, 0);
    } else {
      #pragma unroll
      for (int i = 0; i < 2; ++i) {
        int c   = tid + 256 * i;        // 0..511 8-elem chunks
        int row = c >> 2;
        int col = (c & 3) * 8;
        async_copy16(&As[row][col], Ab + (m0 + row) * lda + k0 + col);
      }
    }
    // ---- stage B tile (128 n x 32 k) via async global->LDS ----
    #pragma unroll
    for (int i = 0; i < 2; ++i) {
      int c   = tid + 256 * i;
      int row = c >> 2;
      int col = (c & 3) * 8;
      async_copy16(&Bs[row][col], Bp + (n0 + row) * ldb + k0 + col);
    }
    wait_async();        // this wave's async LDS writes have landed
    __syncthreads();     // all waves' tiles visible

    // ---- A fragments (ISA 16-bit A 16x32 layout: K pairs at hl*8 and 16+hl*8) ----
    v16bf afrag[2];
    #pragma unroll
    for (int mt = 0; mt < 2; ++mt) {
      const __bf16* ar = &As[wm * 32 + mt * 16 + lrow][0];
      const v8bf a0 = *(const v8bf*)(ar + hl * 8);
      const v8bf a1 = *(const v8bf*)(ar + 16 + hl * 8);
      afrag[mt] = __builtin_shufflevector(a0, a1, 0,1,2,3,4,5,6,7,8,9,10,11,12,13,14,15);
    }
    // ---- B fragments (K pairs at hl*8 and 16+hl*8, matching A) + WMMA ----
    #pragma unroll
    for (int nt = 0; nt < 4; ++nt) {
      const __bf16* br = &Bs[wn * 64 + nt * 16 + lrow][0];
      const v8bf b0 = *(const v8bf*)(br + hl * 8);
      const v8bf b1 = *(const v8bf*)(br + 16 + hl * 8);
      v16bf bfrag = __builtin_shufflevector(b0, b1, 0,1,2,3,4,5,6,7,8,9,10,11,12,13,14,15);
      #pragma unroll
      for (int mt = 0; mt < 2; ++mt) {
        acc[mt][nt] = __builtin_amdgcn_wmma_f32_16x16x32_bf16(
            false, afrag[mt], false, bfrag, (short)0, acc[mt][nt], false, false);
      }
    }
    __syncthreads();   // LDS reuse barrier (before next tile's async writes)
  }

  // ---- epilogue (C layout: VGPR r -> row r + 8*hl, col = lane&15) ----
  const long Ntot = (long)gridDim.x * 128;
  #pragma unroll
  for (int mt = 0; mt < 2; ++mt) {
    #pragma unroll
    for (int nt = 0; nt < 4; ++nt) {
      #pragma unroll
      for (int r = 0; r < 8; ++r) {
        long m = m0 + wm * 32 + mt * 16 + r + 8 * hl;
        long n = n0 + wn * 64 + nt * 16 + lrow;
        float v = acc[mt][nt][r] * scale;
        if (bias)   v += bias[n];
        if (addmat) v += addmat[m * ld_add + n];
        if (c_is_f32) {
          ((float*)Cv)[z * bsC + m * ldc + n] = v;
        } else {
          __bf16* C = (__bf16*)Cv + z * bsC;
          long idx;
          if (c_trans) {
            long bb = m / trans_S, s = m % trans_S;
            idx = bb * Ntot * trans_S + n * (long)trans_S + s;
          } else {
            idx = m * ldc + n;
          }
          C[idx] = f2bf(v);
        }
      }
    }
  }
}

// ---------------- row softmax: f32 scores[row][2048] -> bf16 probs ----------------
__global__ __launch_bounds__(256)
void softmax_kernel(const float* __restrict__ S, __bf16* __restrict__ P) {
  const long row = blockIdx.x;
  const float* s = S + row * (long)SL;
  __bf16* p = P + row * (long)SL;
  const int tid = threadIdx.x;
  __shared__ float red[8];

  float v[8]; float mx = -3.0e38f;
  #pragma unroll
  for (int i = 0; i < 8; ++i) { v[i] = s[tid + 256 * i]; mx = fmaxf(mx, v[i]); }
  mx = waveMax(mx);
  if ((tid & 31) == 0) red[tid >> 5] = mx;
  __syncthreads();
  if (tid == 0) { float m2 = red[0]; for (int i = 1; i < 8; ++i) m2 = fmaxf(m2, red[i]); red[0] = m2; }
  __syncthreads();
  mx = red[0];

  float sum = 0.0f;
  #pragma unroll
  for (int i = 0; i < 8; ++i) { v[i] = __expf(v[i] - mx); sum += v[i]; }
  sum = waveSum(sum);
  __syncthreads();
  if ((tid & 31) == 0) red[tid >> 5] = sum;
  __syncthreads();
  if (tid == 0) { float t = 0.0f; for (int i = 0; i < 8; ++i) t += red[i]; red[0] = t; }
  __syncthreads();
  const float inv = 1.0f / red[0];
  #pragma unroll
  for (int i = 0; i < 8; ++i) p[tid + 256 * i] = f2bf(v[i] * inv);
}

// ---------------- layernorm over last dim (1024) ----------------
__global__ __launch_bounds__(256)
void layernorm_kernel(const float* __restrict__ X, const float* __restrict__ gamma,
                      const float* __restrict__ beta, float* __restrict__ out) {
  const long row = blockIdx.x;
  const float* x = X + row * (long)HD;
  float* o = out + row * (long)HD;
  const int tid = threadIdx.x;
  __shared__ float red[8];

  float v[4]; float s = 0.0f;
  #pragma unroll
  for (int i = 0; i < 4; ++i) { v[i] = x[tid + 256 * i]; s += v[i]; }
  s = waveSum(s);
  if ((tid & 31) == 0) red[tid >> 5] = s;
  __syncthreads();
  if (tid == 0) { float t = 0.0f; for (int i = 0; i < 8; ++i) t += red[i]; red[0] = t; }
  __syncthreads();
  const float mu = red[0] * (1.0f / HD);

  float s2 = 0.0f;
  #pragma unroll
  for (int i = 0; i < 4; ++i) { float d = v[i] - mu; s2 += d * d; }
  s2 = waveSum(s2);
  __syncthreads();
  if ((tid & 31) == 0) red[tid >> 5] = s2;
  __syncthreads();
  if (tid == 0) { float t = 0.0f; for (int i = 0; i < 8; ++i) t += red[i]; red[0] = t; }
  __syncthreads();
  const float rs = rsqrtf(red[0] * (1.0f / HD) + 1e-5f);

  #pragma unroll
  for (int i = 0; i < 4; ++i) {
    int c = tid + 256 * i;
    o[c] = (v[i] - mu) * rs * gamma[c] + beta[c];
  }
}

// ---------------- host side ----------------
extern "C" void kernel_launch(void* const* d_in, const int* in_sizes, int n_in,
                              void* d_out, int out_size, void* d_ws, size_t ws_size,
                              hipStream_t stream) {
  (void)in_sizes; (void)n_in; (void)out_size; (void)ws_size;
  const float* abs_enc = (const float*)d_in[0];
  const float* hidden  = (const float*)d_in[1];
  const float* Wq = (const float*)d_in[2];
  const float* bq = (const float*)d_in[3];
  const float* Wk = (const float*)d_in[4];
  const float* bk = (const float*)d_in[5];
  const float* Wv = (const float*)d_in[6];
  const float* bv = (const float*)d_in[7];
  const float* Wd = (const float*)d_in[8];
  const float* bd = (const float*)d_in[9];
  const float* gamma = (const float*)d_in[10];
  const float* beta  = (const float*)d_in[11];
  float* out = (float*)d_out;

  char* ws = (char*)d_ws;
  const size_t szAct = (size_t)BSROWS * HD * sizeof(uint16_t);          // 33.5 MB
  const size_t szPrb = (size_t)NBATCH * SL * SL * sizeof(uint16_t);     // 67 MB
  const size_t szScr = (size_t)NBATCH * SL * SL * sizeof(float);        // 134 MB
  const size_t szW   = (size_t)HD * HD * sizeof(uint16_t);              // 2 MB

  __bf16* Qb  = (__bf16*)(ws);
  __bf16* Kb  = (__bf16*)(ws + 1 * szAct);
  __bf16* Vt  = (__bf16*)(ws + 2 * szAct);   // transposed per-batch [HD][SL]
  __bf16* Cb  = (__bf16*)(ws + 3 * szAct);
  __bf16* Pb  = (__bf16*)(ws + 4 * szAct);
  float*  Sc  = (float*) (ws + 4 * szAct + szPrb);   // scores; reused as pre-LN tmp
  __bf16* WqT = (__bf16*)(ws + 4 * szAct + szPrb + szScr);
  __bf16* WkT = (__bf16*)((char*)WqT + szW);
  __bf16* WvT = (__bf16*)((char*)WkT + szW);
  __bf16* WdT = (__bf16*)((char*)WvT + szW);

  // 1. weights -> bf16, transposed
  wtrans_kernel<<<HD * HD / 256, 256, 0, stream>>>(Wq, WqT);
  wtrans_kernel<<<HD * HD / 256, 256, 0, stream>>>(Wk, WkT);
  wtrans_kernel<<<HD * HD / 256, 256, 0, stream>>>(Wv, WvT);
  wtrans_kernel<<<HD * HD / 256, 256, 0, stream>>>(Wd, WdT);

  const dim3 blk(256);
  // 2. projections: Q = X Wq + bq ; K' = X Wk + bk + abs ; V' = X Wv + bv + abs (stored transposed)
  gemm_bf16_kernel<<<dim3(HD / 128, BSROWS / 128, 1), blk, 0, stream>>>(
      hidden, 1, HD, 0, WqT, HD, 0, bq, nullptr, 0, 1.0f, Qb, 0, HD, 0, 0, 0, HD);
  gemm_bf16_kernel<<<dim3(HD / 128, BSROWS / 128, 1), blk, 0, stream>>>(
      hidden, 1, HD, 0, WkT, HD, 0, bk, abs_enc, HD, 1.0f, Kb, 0, HD, 0, 0, 0, HD);
  gemm_bf16_kernel<<<dim3(HD / 128, BSROWS / 128, 1), blk, 0, stream>>>(
      hidden, 1, HD, 0, WvT, HD, 0, bv, abs_enc, HD, 1.0f, Vt, 0, HD, 0, 1, SL, HD);

  // 3. scores = Q K'^T / 32  (batched; B operand rows = K rows)
  gemm_bf16_kernel<<<dim3(SL / 128, SL / 128, NBATCH), blk, 0, stream>>>(
      Qb, 0, HD, (long)SL * HD, Kb, HD, (long)SL * HD, nullptr, nullptr, 0,
      0.03125f, Sc, 1, SL, (long)SL * SL, 0, 0, HD);

  // 4. softmax rows -> bf16 probs
  softmax_kernel<<<BSROWS, blk, 0, stream>>>(Sc, Pb);

  // 5. ctx = P V'  (B operand rows = Vt rows, i.e. columns of V')
  gemm_bf16_kernel<<<dim3(HD / 128, SL / 128, NBATCH), blk, 0, stream>>>(
      Pb, 0, SL, (long)SL * SL, Vt, SL, (long)HD * SL, nullptr, nullptr, 0,
      1.0f, Cb, 0, HD, (long)SL * HD, 0, 0, SL);

  // 6. t = ctx Wd + bd + hidden  (f32, into scores region)
  float* Tmp = Sc;
  gemm_bf16_kernel<<<dim3(HD / 128, BSROWS / 128, 1), blk, 0, stream>>>(
      Cb, 0, HD, 0, WdT, HD, 0, bd, hidden, HD, 1.0f, Tmp, 1, HD, 0, 0, 0, HD);

  // 7. layernorm -> output
  layernorm_kernel<<<BSROWS, blk, 0, stream>>>(Tmp, gamma, beta, out);
}
